// Classifer_53584011985482
// MI455X (gfx1250) — 8-bit matrix operand order repaired and checked by reading, compiled, not run on hardware
//
#include <hip/hip_runtime.h>
#include <stdint.h>

#define BATCH 8
#define CCH   128
#define NCLS  19
#define HH    128
#define WW    256
#define TILE  64
#define LSTR  160

typedef int v8i __attribute__((ext_vector_type(8)));

__device__ __forceinline__ v8i wmma_iu8(v8i a, v8i b, v8i c) {
  return __builtin_amdgcn_wmma_i32_16x16x64_iu8(
       true, a,  false, b, c,
       false,  false);
}

__device__ __forceinline__ float clamp_round_q4(float v) {
  v = rintf(v);
  return fminf(fmaxf(v, 0.f), 15.f);
}

__global__ __launch_bounds__(256) void k0_prep(
    const float* __restrict__ dw1_w, const float* __restrict__ pw1_w,
    const float* __restrict__ dw2_w, const float* __restrict__ pw2_w,
    const float* __restrict__ cls_w, const float* __restrict__ cls_b,
    const float* __restrict__ bn1g, const float* __restrict__ bn1b,
    const float* __restrict__ bn1m, const float* __restrict__ bn1v,
    const float* __restrict__ bn2g, const float* __restrict__ bn2b,
    const float* __restrict__ bn2m, const float* __restrict__ bn2v,
    const float* __restrict__ bn3g, const float* __restrict__ bn3b,
    const float* __restrict__ bn3m, const float* __restrict__ bn3v,
    const float* __restrict__ bn4g, const float* __restrict__ bn4b,
    const float* __restrict__ bn4m, const float* __restrict__ bn4v,
    const float* __restrict__ s1p, const float* __restrict__ s2p,
    const float* __restrict__ s3p, const float* __restrict__ s4p,
    float* __restrict__ dw1wq, float* __restrict__ alpha1, float* __restrict__ beta1,
    int8_t* __restrict__ pw1i,  float* __restrict__ alpha2, float* __restrict__ beta2,
    int8_t* __restrict__ dw2i,  float* __restrict__ alpha3, float* __restrict__ beta3,
    int8_t* __restrict__ pw2i,  float* __restrict__ alpha4, float* __restrict__ beta4,
    int8_t* __restrict__ clsi,  float* __restrict__ gammav, float* __restrict__ bqv)
{
  const int t = threadIdx.x;
  const float s1 = *s1p, s2 = *s2p, s3 = *s3p, s4 = *s4p;
  const float EPS = 1e-5f;

  if (t < CCH) {
    const int c = t;
    {
      float mx = 0.f;
      for (int k = 0; k < 9; ++k) mx = fmaxf(mx, fabsf(dw1_w[c*9 + k]));
      float s = fmaxf(mx / 7.f, 1e-8f);
      for (int k = 0; k < 9; ++k)
        dw1wq[c*9 + k] = fminf(fmaxf(rintf(dw1_w[c*9 + k] / s), -7.f), 7.f) * s;
      float bs = bn1g[c] / sqrtf(bn1v[c] + EPS);
      alpha1[c] = bs / s1;
      beta1[c]  = (bn1b[c] - bn1m[c] * bs) / s1;
    }
    {
      float mx = 0.f;
      for (int i = 0; i < CCH; ++i) mx = fmaxf(mx, fabsf(pw1_w[c*CCH + i]));
      float s = fmaxf(mx / 7.f, 1e-8f);
      for (int i = 0; i < CCH; ++i)
        pw1i[c*CCH + i] = (int8_t)fminf(fmaxf(rintf(pw1_w[c*CCH + i] / s), -7.f), 7.f);
      float bs = bn2g[c] / sqrtf(bn2v[c] + EPS);
      alpha2[c] = s * s1 * bs / s2;
      beta2[c]  = (bn2b[c] - bn2m[c] * bs) / s2;
    }
    {
      float mx = 0.f;
      for (int k = 0; k < 9; ++k) mx = fmaxf(mx, fabsf(dw2_w[c*9 + k]));
      float s = fmaxf(mx / 7.f, 1e-8f);
      for (int k = 0; k < 9; ++k)
        dw2i[k*CCH + c] = (int8_t)fminf(fmaxf(rintf(dw2_w[c*9 + k] / s), -7.f), 7.f);
      float bs = bn3g[c] / sqrtf(bn3v[c] + EPS);
      alpha3[c] = s2 * s * bs / s3;
      beta3[c]  = (bn3b[c] - bn3m[c] * bs) / s3;
    }
    {
      float mx = 0.f;
      for (int i = 0; i < CCH; ++i) mx = fmaxf(mx, fabsf(pw2_w[c*CCH + i]));
      float s = fmaxf(mx / 7.f, 1e-8f);
      for (int i = 0; i < CCH; ++i)
        pw2i[c*CCH + i] = (int8_t)fminf(fmaxf(rintf(pw2_w[c*CCH + i] / s), -7.f), 7.f);
      float bs = bn4g[c] / sqrtf(bn4v[c] + EPS);
      alpha4[c] = s * s3 * bs / s4;
      beta4[c]  = (bn4b[c] - bn4m[c] * bs) / s4;
    }
  }
  if (t < 32) {
    const int o = t;
    if (o < NCLS) {
      float mx = 0.f;
      for (int i = 0; i < CCH; ++i) mx = fmaxf(mx, fabsf(cls_w[o*CCH + i]));
      float s = fmaxf(mx / 7.f, 1e-8f);
      for (int i = 0; i < CCH; ++i)
        clsi[o*CCH + i] = (int8_t)fminf(fmaxf(rintf(cls_w[o*CCH + i] / s), -7.f), 7.f);
      float sb = s4 * s;
      gammav[o] = sb;
      bqv[o]    = rintf(cls_b[o] / sb) * sb;
    } else {
      for (int i = 0; i < CCH; ++i) clsi[o*CCH + i] = 0;
      gammav[o] = 0.f; bqv[o] = 0.f;
    }
  }
}

__device__ __forceinline__ void gemm_tile_q4(
    const uint8_t* __restrict__ src, uint8_t* __restrict__ dst,
    const int8_t* __restrict__ wint, const float* __restrict__ alpha,
    const float* __restrict__ beta, int mtile, int lane)
{
  const int m16 = lane & 15, hi = lane >> 4;
  v8i A0, A1;
#pragma unroll
  for (int v = 0; v < 8; ++v) {
    const int coff = ((v >> 1) << 4) + ((v & 1) << 2) + (hi << 3);
    const int row  = (mtile * 16 + m16) * CCH;
    A0[v] = *(const int*)(wint + row + coff);
    A1[v] = *(const int*)(wint + row + 64 + coff);
  }
#pragma unroll
  for (int nt = 0; nt < 4; ++nt) {
    const int p = nt * 16 + m16;
    v8i B0, B1;
#pragma unroll
    for (int v = 0; v < 8; ++v) {
      const int coff = ((v >> 1) << 4) + ((v & 1) << 2) + (hi << 3);
      B0[v] = *(const int*)&src[p * LSTR + coff];
      B1[v] = *(const int*)&src[p * LSTR + 64 + coff];
    }
    v8i Cacc = {0, 0, 0, 0, 0, 0, 0, 0};
    Cacc = wmma_iu8(A0, B0, Cacc);
    Cacc = wmma_iu8(A1, B1, Cacc);
#pragma unroll
    for (int r = 0; r < 8; ++r) {
      const int o = mtile * 16 + hi * 8 + r;
      dst[p * LSTR + o] = (uint8_t)clamp_round_q4((float)Cacc[r] * alpha[o] + beta[o]);
    }
  }
}

__global__ __launch_bounds__(256) void k1_dw1_pw1(
    const float* __restrict__ x, const float* __restrict__ dw1wq,
    const float* __restrict__ alpha1, const float* __restrict__ beta1,
    const int8_t* __restrict__ pw1i,
    const float* __restrict__ alpha2, const float* __restrict__ beta2,
    uint8_t* __restrict__ a2)
{
  __shared__ uint8_t a1s[TILE * LSTR];
  __shared__ uint8_t a2s[TILE * LSTR];
  const int t  = threadIdx.x;
  const int w0 = blockIdx.x * TILE;
  const int h  = blockIdx.y;
  const int b  = blockIdx.z;

  {
    const int p  = t & 63;
    const int cb = (t >> 6) * 32;
    const int wg = w0 + p;
#pragma unroll 2
    for (int i = 0; i < 32; ++i) {
      const int c = cb + i;
      const float* wq = dw1wq + c * 9;
      const float* xc = x + ((size_t)(b * CCH + c) * HH) * WW;
      float acc = 0.f;
#pragma unroll
      for (int dh = -1; dh <= 1; ++dh) {
        const int hh = h + dh;
        if (hh < 0 || hh >= HH) continue;
#pragma unroll
        for (int dw = -1; dw <= 1; ++dw) {
          const int ww = wg + dw;
          if (ww < 0 || ww >= WW) continue;
          acc += xc[hh * WW + ww] * wq[(dh + 1) * 3 + (dw + 1)];
        }
      }
      a1s[p * LSTR + c] = (uint8_t)clamp_round_q4(acc * alpha1[c] + beta1[c]);
    }
  }
  __syncthreads();

  gemm_tile_q4(a1s, a2s, pw1i, alpha2, beta2,  t >> 5,  t & 31);
  __syncthreads();

  {
    const int pp = t >> 2, sub = t & 3;
    uint8_t* dst = a2 + ((size_t)((b * HH + h) * WW) + w0 + pp) * CCH;
    *(uint4*)(dst + sub * 16)      = *(const uint4*)&a2s[pp * LSTR + sub * 16];
    *(uint4*)(dst + 64 + sub * 16) = *(const uint4*)&a2s[pp * LSTR + 64 + sub * 16];
  }
}

__global__ __launch_bounds__(256) void k2_dw2_pw2_cls(
    const uint8_t* __restrict__ a2,
    const int8_t* __restrict__ dw2i,
    const float* __restrict__ alpha3, const float* __restrict__ beta3,
    const int8_t* __restrict__ pw2i,
    const float* __restrict__ alpha4, const float* __restrict__ beta4,
    const int8_t* __restrict__ clsi,
    const float* __restrict__ gammav, const float* __restrict__ bqv,
    float* __restrict__ out)
{
  __shared__ uint8_t a3s[TILE * LSTR];
  __shared__ uint8_t a4s[TILE * LSTR];
  const int t  = threadIdx.x;
  const int w0 = blockIdx.x * TILE;
  const int h  = blockIdx.y;
  const int b  = blockIdx.z;
  const int lane = t & 31, wv = t >> 5;

  {
    const int p  = t & 63;
    const int cb = (t >> 6) * 32;
    int acc[32];
#pragma unroll
    for (int j = 0; j < 32; ++j) acc[j] = 0;
#pragma unroll
    for (int tap = 0; tap < 9; ++tap) {
      const int dh = tap / 3 - 1, dw = tap % 3 - 1;
      const int hh = h + dh, ww = w0 + p + dw;
      if (hh < 0 || hh >= HH || ww < 0 || ww >= WW) continue;
      const uint8_t* ar = a2 + ((size_t)((b * HH + hh) * WW) + ww) * CCH + cb;
      const int8_t*  wr = dw2i + tap * CCH + cb;
#pragma unroll
      for (int k = 0; k < 8; ++k) {
        const unsigned av = *(const unsigned*)(ar + k * 4);
        const unsigned wu = *(const unsigned*)(wr + k * 4);
#pragma unroll
        for (int j = 0; j < 4; ++j)
          acc[k * 4 + j] += (int)((av >> (8 * j)) & 0xFFu) *
                            (int)(int8_t)(wu >> (8 * j));
      }
    }
#pragma unroll
    for (int j = 0; j < 32; ++j) {
      const int c = cb + j;
      a3s[p * LSTR + c] = (uint8_t)clamp_round_q4((float)acc[j] * alpha3[c] + beta3[c]);
    }
  }
  __syncthreads();

  gemm_tile_q4(a3s, a4s, pw2i, alpha4, beta4,  wv, lane);
  __syncthreads();

  {
    const int mtile = wv >> 2;
    const int ntile = wv & 3;
    const int m16 = lane & 15, hi = lane >> 4;
    v8i A0, A1;
#pragma unroll
    for (int v = 0; v < 8; ++v) {
      const int coff = ((v >> 1) << 4) + ((v & 1) << 2) + (hi << 3);
      const int row  = (mtile * 16 + m16) * CCH;
      A0[v] = *(const int*)(clsi + row + coff);
      A1[v] = *(const int*)(clsi + row + 64 + coff);
    }
    const int p = ntile * 16 + m16;
    v8i B0, B1;
#pragma unroll
    for (int v = 0; v < 8; ++v) {
      const int coff = ((v >> 1) << 4) + ((v & 1) << 2) + (hi << 3);
      B0[v] = *(const int*)&a4s[p * LSTR + coff];
      B1[v] = *(const int*)&a4s[p * LSTR + 64 + coff];
    }
    v8i Cacc = {0, 0, 0, 0, 0, 0, 0, 0};
    Cacc = wmma_iu8(A0, B0, Cacc);
    Cacc = wmma_iu8(A1, B1, Cacc);
#pragma unroll
    for (int r = 0; r < 8; ++r) {
      const int o = mtile * 16 + hi * 8 + r;
      if (o < NCLS)
        out[(((size_t)b * NCLS + o) * HH + h) * WW + w0 + p] =
            (float)Cacc[r] * gammav[o] + bqv[o];
    }
  }
}

static constexpr size_t OFF_A2     = 0;
static constexpr size_t OFF_DW1WQ  = 33554432;
static constexpr size_t OFF_ALPHA1 = OFF_DW1WQ + 8192;
static constexpr size_t OFF_BETA1  = OFF_ALPHA1 + 512;
static constexpr size_t OFF_PW1I   = OFF_BETA1 + 512;
static constexpr size_t OFF_ALPHA2 = OFF_PW1I + 16384;
static constexpr size_t OFF_BETA2  = OFF_ALPHA2 + 512;
static constexpr size_t OFF_DW2I   = OFF_BETA2 + 512;
static constexpr size_t OFF_ALPHA3 = OFF_DW2I + 2048;
static constexpr size_t OFF_BETA3  = OFF_ALPHA3 + 512;
static constexpr size_t OFF_PW2I   = OFF_BETA3 + 512;
static constexpr size_t OFF_ALPHA4 = OFF_PW2I + 16384;
static constexpr size_t OFF_BETA4  = OFF_ALPHA4 + 512;
static constexpr size_t OFF_CLSI   = OFF_BETA4 + 512;
static constexpr size_t OFF_GAMMA  = OFF_CLSI + 4096;
static constexpr size_t OFF_BQ     = OFF_GAMMA + 512;

extern "C" void kernel_launch(void* const* d_in, const int* in_sizes, int n_in,
                              void* d_out, int out_size, void* d_ws, size_t ws_size,
                              hipStream_t stream) {
  (void)in_sizes; (void)n_in; (void)out_size; (void)ws_size;
  const float* x     = (const float*)d_in[0];
  const float* dw1_w = (const float*)d_in[1];
  const float* pw1_w = (const float*)d_in[2];
  const float* dw2_w = (const float*)d_in[3];
  const float* pw2_w = (const float*)d_in[4];
  const float* cls_w = (const float*)d_in[5];
  const float* cls_b = (const float*)d_in[6];
  const float* bn[16];
  for (int i = 0; i < 16; ++i) bn[i] = (const float*)d_in[7 + i];
  const float* s1 = (const float*)d_in[23];
  const float* s2 = (const float*)d_in[24];
  const float* s3 = (const float*)d_in[25];
  const float* s4 = (const float*)d_in[26];

  char* ws = (char*)d_ws;
  uint8_t* a2    = (uint8_t*)(ws + OFF_A2);
  float* dw1wq   = (float*)(ws + OFF_DW1WQ);
  float* alpha1  = (float*)(ws + OFF_ALPHA1);
  float* beta1   = (float*)(ws + OFF_BETA1);
  int8_t* pw1i   = (int8_t*)(ws + OFF_PW1I);
  float* alpha2  = (float*)(ws + OFF_ALPHA2);
  float* beta2   = (float*)(ws + OFF_BETA2);
  int8_t* dw2i   = (int8_t*)(ws + OFF_DW2I);
  float* alpha3  = (float*)(ws + OFF_ALPHA3);
  float* beta3   = (float*)(ws + OFF_BETA3);
  int8_t* pw2i   = (int8_t*)(ws + OFF_PW2I);
  float* alpha4  = (float*)(ws + OFF_ALPHA4);
  float* beta4   = (float*)(ws + OFF_BETA4);
  int8_t* clsi   = (int8_t*)(ws + OFF_CLSI);
  float* gammav  = (float*)(ws + OFF_GAMMA);
  float* bqv     = (float*)(ws + OFF_BQ);

  k0_prep<<<1, 256, 0, stream>>>(
      dw1_w, pw1_w, dw2_w, pw2_w, cls_w, cls_b,
      bn[0], bn[1], bn[2], bn[3], bn[4], bn[5], bn[6], bn[7],
      bn[8], bn[9], bn[10], bn[11], bn[12], bn[13], bn[14], bn[15],
      s1, s2, s3, s4,
      dw1wq, alpha1, beta1, pw1i, alpha2, beta2,
      dw2i, alpha3, beta3, pw2i, alpha4, beta4, clsi, gammav, bqv);

  dim3 grid(WW / TILE, HH, BATCH);
  k1_dw1_pw1<<<grid, 256, 0, stream>>>(x, dw1wq, alpha1, beta1,
                                       pw1i, alpha2, beta2, a2);
  k2_dw2_pw2_cls<<<grid, 256, 0, stream>>>(a2, dw2i, alpha3, beta3,
                                           pw2i, alpha4, beta4,
                                           clsi, gammav, bqv, (float*)d_out);
}
